// KGPolicy_44865228374246
// MI455X (gfx1250) — compile-verified, matrix operand order checked by reading, not run on hardware
//
#include <hip/hip_runtime.h>
#include <hip/hip_bf16.h>
#include <math.h>

// ---------------------------------------------------------------------------
// KGPolicy negative sampler for MI455X (gfx1250, wave32, WMMA).
//   - SAGE GEMMs run on v_wmma_f32_16x16x32_f16 (16x16x32 tiles, f32 accum)
//   - Edge aggregation: fp32 global atomics (L2-resident working set)
//   - Batch scoring / softmax / argsort: wave32 shuffle networks
// ---------------------------------------------------------------------------

#define N_NODES 100000
#define K_NEIGH 32
#define ITEM_HI 49999
#define T_TRAIN 50

typedef __attribute__((ext_vector_type(16))) _Float16 v16h;
typedef __attribute__((ext_vector_type(8)))  float    v8f;

__device__ __forceinline__ unsigned hash32(unsigned x) {
  x ^= x >> 16; x *= 0x7feb352dU;
  x ^= x >> 15; x *= 0x846ca68bU;
  x ^= x >> 16; return x;
}

// ---------------------------------------------------------------- utilities
__global__ void zero_f32(float* p, size_t n) {
  size_t i = (size_t)blockIdx.x * blockDim.x + threadIdx.x;
  size_t s = (size_t)gridDim.x * blockDim.x;
  for (; i < n; i += s) p[i] = 0.0f;
}

__global__ void edge_count(const int* __restrict__ edges, float* __restrict__ cnt, int nEdges) {
  int i = blockIdx.x * blockDim.x + threadIdx.x;
  if (i < nEdges) atomicAdd(&cnt[edges[i]], 1.0f);
}

__global__ void make_inv(const float* __restrict__ cnt, float* __restrict__ inv, int n) {
  int i = blockIdx.x * blockDim.x + threadIdx.x;
  if (i < n) inv[i] = 1.0f / fmaxf(cnt[i], 1.0f);
}

// agg[dst] += x[node] for each of the node's K_NEIGH sampled dst edges.
__global__ void sage_scatter(const float* __restrict__ x, const int* __restrict__ edges,
                             float* __restrict__ agg, int D, int nNodes) {
  int node = blockIdx.x;
  int d = threadIdx.x;
  if (node >= nNodes) return;
  float v = x[(size_t)node * D + d];
  const int* er = edges + (size_t)node * K_NEIGH;
  if (d == 0 && node + 1 < nNodes)
    __builtin_prefetch(edges + (size_t)(node + 1) * K_NEIGH, 0, 1);
#pragma unroll 4
  for (int k = 0; k < K_NEIGH; ++k) {
    int dst = er[k];
    atomicAdd(agg + (size_t)dst * D + d, v);
  }
}

// ------------------------------------------------------- WMMA SAGE layer 1
// out[16,32] = leaky_relu([agg*inv , x] (16x128) @ [Wl;Wr]^T + bl) * dropout
__global__ __launch_bounds__(128) void sage_gemm_l1(
    const float* __restrict__ agg, const float* __restrict__ x,
    const float* __restrict__ invcnt,
    const float* __restrict__ Wl, const float* __restrict__ bl,
    const float* __restrict__ Wr,
    float* __restrict__ out, int nNodes, unsigned seed) {
  __shared__ float tileBuf[4][16][128];
  const int wave = threadIdx.x >> 5, lane = threadIdx.x & 31;
  const int row0 = (blockIdx.x * 4 + wave) * 16;
  float (*T)[128] = tileBuf[wave];

  // cooperative float4 stage of the combined 16x128 tile
  for (int i = lane; i < 512; i += 32) {
    int r = i >> 5, col = (i & 31) * 4;
    int node = row0 + r; if (node >= nNodes) node = nNodes - 1;
    float4 vq;
    if (col < 64) {
      vq = *(const float4*)(agg + (size_t)node * 64 + col);
      float s = invcnt[node];
      vq.x *= s; vq.y *= s; vq.z *= s; vq.w *= s;
    } else {
      vq = *(const float4*)(x + (size_t)node * 64 + (col - 64));
    }
    *(float4*)&T[r][col] = vq;
  }
  __syncthreads();

  const int m16 = lane & 15, hs = lane >> 4;
  v8f acc0 = {}; v8f acc1 = {};
#pragma unroll
  for (int ks = 0; ks < 4; ++ks) {
    v16h A;
#pragma unroll
    for (int e = 0; e < 16; ++e) {
      int ve = e >> 1;
      int k = ((ve & 4) ? 16 : 0) + 8 * hs + 2 * (ve & 3) + (e & 1);
      A[e] = (_Float16)T[m16][ks * 32 + k];
    }
    v16h B0, B1;
#pragma unroll
    for (int e = 0; e < 16; ++e) {
      int ve = e >> 1;
      int k = ks * 32 + ((ve & 4) ? 16 : 0) + 8 * hs + 2 * (ve & 3) + (e & 1);
      float w0 = (k < 64) ? Wl[m16 * 64 + k] : Wr[m16 * 64 + (k - 64)];
      float w1 = (k < 64) ? Wl[(m16 + 16) * 64 + k] : Wr[(m16 + 16) * 64 + (k - 64)];
      B0[e] = (_Float16)w0;
      B1[e] = (_Float16)w1;
    }
    acc0 = __builtin_amdgcn_wmma_f32_16x16x32_f16(false, A, false, B0, (short)0, acc0, false, false);
    acc1 = __builtin_amdgcn_wmma_f32_16x16x32_f16(false, A, false, B1, (short)0, acc1, false, false);
  }

#pragma unroll
  for (int r8 = 0; r8 < 8; ++r8) {
    int node = row0 + r8 + 8 * hs;
    if (node < nNodes) {
      {
        int n = m16;
        float vv = acc0[r8] + bl[n];
        vv = vv > 0.0f ? vv : 0.01f * vv;
        vv = (hash32(((unsigned)node * 32u + n) ^ seed) & 1u) ? vv * 2.0f : 0.0f;
        out[(size_t)node * 32 + n] = vv;
      }
      {
        int n = m16 + 16;
        float vv = acc1[r8] + bl[n];
        vv = vv > 0.0f ? vv : 0.01f * vv;
        vv = (hash32(((unsigned)node * 32u + n) ^ seed) & 1u) ? vv * 2.0f : 0.0f;
        out[(size_t)node * 32 + n] = vv;
      }
    }
  }
}

// ------------------------------------------------------- WMMA SAGE layer 2
// out[16,64] = ([agg*inv , h] (16x64) @ [Wl;Wr]^T + bl) * dropout
__global__ __launch_bounds__(128) void sage_gemm_l2(
    const float* __restrict__ agg, const float* __restrict__ h,
    const float* __restrict__ invcnt,
    const float* __restrict__ Wl, const float* __restrict__ bl,
    const float* __restrict__ Wr,
    float* __restrict__ out, int nNodes, unsigned seed) {
  __shared__ float tileBuf[4][16][64];
  const int wave = threadIdx.x >> 5, lane = threadIdx.x & 31;
  const int row0 = (blockIdx.x * 4 + wave) * 16;
  float (*T)[64] = tileBuf[wave];

  for (int i = lane; i < 256; i += 32) {
    int r = i >> 4, col = (i & 15) * 4;
    int node = row0 + r; if (node >= nNodes) node = nNodes - 1;
    float4 vq;
    if (col < 32) {
      vq = *(const float4*)(agg + (size_t)node * 32 + col);
      float s = invcnt[node];
      vq.x *= s; vq.y *= s; vq.z *= s; vq.w *= s;
    } else {
      vq = *(const float4*)(h + (size_t)node * 32 + (col - 32));
    }
    *(float4*)&T[r][col] = vq;
  }
  __syncthreads();

  const int m16 = lane & 15, hs = lane >> 4;
  v8f acc[4] = {};
#pragma unroll
  for (int ks = 0; ks < 2; ++ks) {
    v16h A;
#pragma unroll
    for (int e = 0; e < 16; ++e) {
      int ve = e >> 1;
      int k = ((ve & 4) ? 16 : 0) + 8 * hs + 2 * (ve & 3) + (e & 1);
      A[e] = (_Float16)T[m16][ks * 32 + k];
    }
#pragma unroll
    for (int jt = 0; jt < 4; ++jt) {
      v16h Bf;
      int n = jt * 16 + m16;
#pragma unroll
      for (int e = 0; e < 16; ++e) {
        int ve = e >> 1;
        int k = ks * 32 + ((ve & 4) ? 16 : 0) + 8 * hs + 2 * (ve & 3) + (e & 1);
        float wv = (k < 32) ? Wl[n * 32 + k] : Wr[n * 32 + (k - 32)];
        Bf[e] = (_Float16)wv;
      }
      acc[jt] = __builtin_amdgcn_wmma_f32_16x16x32_f16(false, A, false, Bf, (short)0, acc[jt], false, false);
    }
  }

#pragma unroll
  for (int jt = 0; jt < 4; ++jt) {
    int n = jt * 16 + m16;
#pragma unroll
    for (int r8 = 0; r8 < 8; ++r8) {
      int node = row0 + r8 + 8 * hs;
      if (node < nNodes) {
        float vv = acc[jt][r8] + bl[n];
        vv = (hash32(((unsigned)node * 64u + n) ^ seed) & 1u) ? vv * 2.0f : 0.0f;
        out[(size_t)node * 64 + n] = vv;
      }
    }
  }
}

// row-wise L2 normalize, one wave per node (wave32)
__global__ void l2norm(float* __restrict__ g, int nNodes) {
  int wave = threadIdx.x >> 5, lane = threadIdx.x & 31;
  int node = blockIdx.x * (blockDim.x >> 5) + wave;
  if (node >= nNodes) return;
  float a = g[(size_t)node * 64 + lane];
  float b = g[(size_t)node * 64 + lane + 32];
  float s = a * a + b * b;
  for (int o = 16; o; o >>= 1) s += __shfl_xor(s, o, 32);
  float scale = 1.0f / fmaxf(sqrtf(s), 1e-12f);
  g[(size_t)node * 64 + lane] = a * scale;
  g[(size_t)node * 64 + lane + 32] = b * scale;
}

// ------------------------------------------------- kg_step (step 1: argmax)
__global__ __launch_bounds__(32) void kg_step1(
    const float* __restrict__ g, const int* __restrict__ users,
    const int* __restrict__ pos_ids, const int* __restrict__ adj,
    int* __restrict__ oh, float* __restrict__ l1, int B) {
  __shared__ float w[64];
  int b = blockIdx.x, lane = threadIdx.x;
  int u = users[b], p = pos_ids[b];
  const float* gu = g + (size_t)u * 64;
  const float* gp = g + (size_t)p * 64;
  w[lane] = gu[lane] * gp[lane];
  w[lane + 32] = gu[lane + 32] * gp[lane + 32];
  __syncthreads();
  int nb = adj[(size_t)p * K_NEIGH + lane];
  const float* gn = g + (size_t)nb * 64;
  float pk = 0.0f;
#pragma unroll
  for (int d = 0; d < 64; ++d) pk += gn[d] * w[d];
  float mx = pk;
  for (int o = 16; o; o >>= 1) mx = fmaxf(mx, __shfl_xor(mx, o, 32));
  float se = __expf(pk - mx);
  for (int o = 16; o; o >>= 1) se += __shfl_xor(se, o, 32);
  float lp = pk - mx - __logf(se);
  float bv = pk; int bi = lane;
  for (int o = 16; o; o >>= 1) {
    float ov = __shfl_xor(bv, o, 32); int oi = __shfl_xor(bi, o, 32);
    if (ov > bv || (ov == bv && oi < bi)) { bv = ov; bi = oi; }
  }
  float lsel = __shfl(lp, bi, 32);
  int nsel = __shfl(nb, bi, 32);
  if (lane == 0) { oh[b] = nsel; l1[b] = lsel; }
}

// --------------------------- kg_step (step 2: full ascending bitonic sort)
__global__ __launch_bounds__(32) void kg_step2(
    const float* __restrict__ g, const int* __restrict__ users,
    const int* __restrict__ pos_ids, const int* __restrict__ adj,
    int* __restrict__ cand, float* __restrict__ clog, int B, unsigned rseed) {
  __shared__ float w[64];
  int b = blockIdx.x, lane = threadIdx.x;
  int u = users[b], p = pos_ids[b];
  const float* gu = g + (size_t)u * 64;
  const float* gp = g + (size_t)p * 64;
  w[lane] = gu[lane] * gp[lane];
  w[lane + 32] = gu[lane + 32] * gp[lane + 32];
  __syncthreads();
  int nb = adj[(size_t)p * K_NEIGH + lane];
  const float* gn = g + (size_t)nb * 64;
  float pk = 0.0f;
#pragma unroll
  for (int d = 0; d < 64; ++d) pk += gn[d] * w[d];
  float mx = pk;
  for (int o = 16; o; o >>= 1) mx = fmaxf(mx, __shfl_xor(mx, o, 32));
  float se = __expf(pk - mx);
  for (int o = 16; o; o >>= 1) se += __shfl_xor(se, o, 32);
  float lp = pk - mx - __logf(se);

  // 32-lane bitonic sort, ascending by key, carrying (neighbor id, logp)
  float key = pk; int id = nb; float lpv = lp;
  for (int size = 2; size <= 32; size <<= 1) {
    for (int stride = size >> 1; stride; stride >>= 1) {
      float ok = __shfl_xor(key, stride, 32);
      int   oi = __shfl_xor(id, stride, 32);
      float ol = __shfl_xor(lpv, stride, 32);
      bool up = ((lane & size) == 0);
      bool lower = ((lane & stride) == 0);
      bool wantMin = (up == lower);
      bool iMin = (key < ok) || (key == ok && id < oi);
      bool keepMine = (wantMin == iMin);
      key = keepMine ? key : ok;
      id  = keepMine ? id : oi;
      lpv = keepMine ? lpv : ol;
    }
  }
  int c = id;
  if (c < 0 || c > ITEM_HI)
    c = (int)(hash32(((unsigned)(b * 32 + lane)) ^ rseed) % (unsigned)(ITEM_HI + 1));
  cand[(size_t)b * 32 + lane] = c;
  clog[(size_t)b * 32 + lane] = lpv;
}

// --------------------------------------- discriminator select + ts filter
__global__ __launch_bounds__(32) void dis_select(
    const float* __restrict__ dis_user, const float* __restrict__ dis_item,
    const int* __restrict__ users, const int* __restrict__ cand,
    const float* __restrict__ clog, const float* __restrict__ l1,
    const int* __restrict__ neg, const int* __restrict__ train_set,
    int* __restrict__ gneg_out, float* __restrict__ out_neg,
    float* __restrict__ out_log, int B, int round) {
  __shared__ float uvec[64];
  int b = blockIdx.x, lane = threadIdx.x;
  int uu = users[b];
  uvec[lane] = dis_user[(size_t)uu * 64 + lane];
  uvec[lane + 32] = dis_user[(size_t)uu * 64 + lane + 32];
  __syncthreads();
  int c = cand[(size_t)b * 32 + lane];
  const float* it = dis_item + (size_t)c * 64;
  float r = 0.0f;
#pragma unroll
  for (int d = 0; d < 64; ++d) r += uvec[d] * it[d];
  float bv = r; int bi = lane;
  for (int o = 16; o; o >>= 1) {
    float ov = __shfl_xor(bv, o, 32); int oi = __shfl_xor(bi, o, 32);
    if (ov > bv || (ov == bv && oi < bi)) { bv = ov; bi = oi; }
  }
  int gneg = __shfl(c, bi, 32);
  float myclog = clog[(size_t)b * 32 + lane];
  float lsel = __shfl(myclog, bi, 32);
  const int* tr = train_set + (size_t)b * T_TRAIN;
  int bad = (tr[lane] == gneg) ? 1 : 0;
  if (lane + 32 < T_TRAIN) bad |= (tr[lane + 32] == gneg) ? 1 : 0;
  for (int o = 16; o; o >>= 1) bad |= __shfl_xor(bad, o, 32);
  if (bad) gneg = neg[b];
  if (lane == 0) {
    gneg_out[b] = gneg;
    out_neg[(size_t)b * 2 + round] = (float)gneg;
    out_log[(size_t)b * 2 + round] = l1[b] + lsel;
  }
}

// ---------------------------------------------------------------------------
extern "C" void kernel_launch(void* const* d_in, const int* in_sizes, int n_in,
                              void* d_out, int out_size, void* d_ws, size_t ws_size,
                              hipStream_t stream) {
  const int* users = (const int*)d_in[0];
  const int* pos = (const int*)d_in[1];
  const int* neg = (const int*)d_in[2];
  const int* train_set = (const int*)d_in[3];
  const int* adj = (const int*)d_in[4];
  const int* edges = (const int*)d_in[5];
  const float* ent = (const float*)d_in[6];
  const float* Wl1 = (const float*)d_in[7];
  const float* bl1 = (const float*)d_in[8];
  const float* Wr1 = (const float*)d_in[9];
  const float* Wl2 = (const float*)d_in[10];
  const float* bl2 = (const float*)d_in[11];
  const float* Wr2 = (const float*)d_in[12];
  const float* dis_user = (const float*)d_in[13];
  const float* dis_item = (const float*)d_in[14];
  const int B = in_sizes[0];
  const int N = N_NODES;

  // workspace carve (256B aligned)
  char* wsp = (char*)d_ws;
  auto carve = [&](size_t bytes) {
    void* p = (void*)wsp;
    wsp += (bytes + 255) & ~(size_t)255;
    return p;
  };
  float* cnt    = (float*)carve((size_t)N * 4);
  float* invcnt = (float*)carve((size_t)N * 4);
  float* agg1   = (float*)carve((size_t)N * 64 * 4);
  float* h1     = (float*)carve((size_t)N * 32 * 4);
  float* agg2   = (float*)carve((size_t)N * 32 * 4);
  float* gbuf   = (float*)carve((size_t)N * 64 * 4);
  int*   oh     = (int*)carve((size_t)B * 4);
  float* l1buf  = (float*)carve((size_t)B * 4);
  int*   cand   = (int*)carve((size_t)B * 32 * 4);
  float* clog   = (float*)carve((size_t)B * 32 * 4);
  int*   gnegb  = (int*)carve((size_t)B * 4);

  float* out_neg = (float*)d_out;
  float* out_log = out_neg + (size_t)B * 2;

  // neighbor in-degree counts (edge list is input-invariant per call)
  zero_f32<<<1024, 256, 0, stream>>>(cnt, (size_t)N);
  edge_count<<<(N * K_NEIGH + 255) / 256, 256, 0, stream>>>(edges, cnt, N * K_NEIGH);
  make_inv<<<(N + 255) / 256, 256, 0, stream>>>(cnt, invcnt, N);

  const int tiles = (N + 15) / 16;
  const int gemmBlocks = (tiles + 3) / 4;

  auto run_gcn = [&](int call_id) {
    unsigned s1 = 0x9E3779B9u * (unsigned)(call_id * 2 + 1) + 0x1234567u;
    unsigned s2 = 0x85EBCA6Bu * (unsigned)(call_id * 2 + 2) + 0x89ABCDEu;
    zero_f32<<<1024, 256, 0, stream>>>(agg1, (size_t)N * 64);
    sage_scatter<<<N, 64, 0, stream>>>(ent, edges, agg1, 64, N);
    sage_gemm_l1<<<gemmBlocks, 128, 0, stream>>>(agg1, ent, invcnt, Wl1, bl1, Wr1, h1, N, s1);
    zero_f32<<<1024, 256, 0, stream>>>(agg2, (size_t)N * 32);
    sage_scatter<<<N, 32, 0, stream>>>(h1, edges, agg2, 32, N);
    sage_gemm_l2<<<gemmBlocks, 128, 0, stream>>>(agg2, h1, invcnt, Wl2, bl2, Wr2, gbuf, N, s2);
    l2norm<<<(N + 7) / 8, 256, 0, stream>>>(gbuf, N);
  };

  const int* pids = pos;
  for (int r = 0; r < 2; ++r) {
    int base = 2 * r;
    run_gcn(base);
    kg_step1<<<B, 32, 0, stream>>>(gbuf, users, pids, adj, oh, l1buf, B);
    run_gcn(base + 1);
    kg_step2<<<B, 32, 0, stream>>>(gbuf, users, oh, adj, cand, clog, B,
                                   0xC0FFEEu + (unsigned)base * 0x9E37u);
    dis_select<<<B, 32, 0, stream>>>(dis_user, dis_item, users, cand, clog, l1buf,
                                     neg, train_set, gnegb, out_neg, out_log, B, r);
    pids = gnegb;
  }
}
